// MFHawkes_74603581931551
// MI455X (gfx1250) — matrix operand order repaired and checked by reading, not compiled, not run on hardware
//
#include <hip/hip_runtime.h>
#include <hip/hip_bf16.h>
#include <math.h>

#define B_DIM 8
#define L_DIM 2048
#define D_DIM 128
#define TM 64            // i-rows per workgroup
#define TJ 64            // j-columns per LDS super-tile
#define LDSPAD 8         // halves of padding per row -> conflict-free ds_load_b128
#define EPS 1e-6f
#define NT 256           // threads per block
#define F4PT (TJ * (D_DIM / 4) / NT)   // float4 fetches per thread per super-tile (=8)

typedef __attribute__((ext_vector_type(16))) _Float16 v16h;
typedef __attribute__((ext_vector_type(8)))  _Float16 v8h;
typedef __attribute__((ext_vector_type(8)))  float    v8f;

__global__ __launch_bounds__(NT) void
mf_hawkes_wmma_kernel(const int*   __restrict__ ids,    // (B,L)
                      const float* __restrict__ times,  // (B,L)
                      const float* __restrict__ maskp,  // (B,L)
                      const float* __restrict__ emb,    // (K,D)
                      const float* __restrict__ utab,   // (K,1)
                      const float* __restrict__ betap,  // scalar
                      float*       __restrict__ out)    // (B,)
{
    // Double-buffered j super-tile; buffer 0 doubles as the i-tile staging
    // area before the pipeline starts (dead after A-fragment extraction).
    __shared__ _Float16 sEj[2][TJ][D_DIM + LDSPAD];
    __shared__ float sWj[2][TJ];   // beta * exp(beta*(t_j - t_ref)) * mask_j
    __shared__ float sTi[TM];
    __shared__ float sU[TM];
    __shared__ float sLam[TM];
    __shared__ float sTref;

    const int wg   = blockIdx.x;           // 0 .. B*(L/TM)-1
    const int b    = wg / (L_DIM / TM);
    const int tile = wg % (L_DIM / TM);
    const int i0   = tile * TM;
    const int tid  = threadIdx.x;
    const int lane = tid & 31;
    const int wave = tid >> 5;

    const float beta = betap[0];

    // ---- stage per-row scalars + i-tile embeddings (into buffer 0) ----
    if (tid < TM) {
        const int ig = i0 + tid;
        const int id = ids[b * L_DIM + ig];
        const float t = times[b * L_DIM + ig];
        sTi[tid]  = t;
        sU[tid]   = utab[id];
        sLam[tid] = 0.0f;
        if (tid == 0) sTref = t;   // reference time: first event of the tile
    }
    for (int e = tid; e < TM * (D_DIM / 4); e += NT) {
        const int r  = e / (D_DIM / 4);
        const int c4 = e % (D_DIM / 4);
        const int id = ids[b * L_DIM + i0 + r];
        const float4 v = ((const float4*)(emb + (size_t)id * D_DIM))[c4];
        _Float16* dst = &sEj[0][r][c4 * 4];
        dst[0] = (_Float16)v.x; dst[1] = (_Float16)v.y;
        dst[2] = (_Float16)v.z; dst[3] = (_Float16)v.w;
    }
    __syncthreads();

    const float tref = sTref;

    // ---- per-wave role ----
    const int mtile = wave & 3;     // which 16-row tile of the 64-row block
    const int jpar  = wave >> 2;    // 0/1: alternate 16-col j-tiles
    const int hi    = lane >> 4;    // half-wave select
    const int n     = lane & 15;    // C-fragment column within tile
    const int mrow  = mtile * 16 + n;

    // A fragments (K = 0..127 in four K=32 chunks), resident in VGPRs.
    v16h afrag[4];
#pragma unroll
    for (int c = 0; c < 4; ++c) {
        v8h* ap = (v8h*)&afrag[c];
        ap[0] = *(const v8h*)&sEj[0][mrow][c * 32 + hi * 8];
        ap[1] = *(const v8h*)&sEj[0][mrow][c * 32 + 16 + hi * 8];
    }

    // i-side exp factor for the 8 C rows this lane owns (M = v + 8*hi),
    // applied once AFTER the whole j sweep.
    float eI[8];
#pragma unroll
    for (int v = 0; v < 8; ++v)
        eI[v] = __expf(-beta * (sTi[mtile * 16 + v + 8 * hi] - tref));
    const int igBase = i0 + mtile * 16 + 8 * hi;

    float racc[8];
#pragma unroll
    for (int v = 0; v < 8; ++v) racc[v] = 0.0f;

    __syncthreads();   // buffer 0 free for reuse as j-buffer

    // ---- pipeline helpers ----
    float4 rj[F4PT];            // in-flight global fetches for next super-tile
    float  tjv = 0.f, mjv = 0.f;

    auto fetch_j = [&](int j0) {            // issue global loads only
#pragma unroll
        for (int k = 0; k < F4PT; ++k) {
            const int e  = tid + k * NT;
            const int r  = e / (D_DIM / 4);
            const int c4 = e % (D_DIM / 4);
            const int id = ids[b * L_DIM + j0 + r];
            rj[k] = ((const float4*)(emb + (size_t)id * D_DIM))[c4];
        }
        if (tid < TJ) {
            tjv = times[b * L_DIM + j0 + tid];
            mjv = maskp[b * L_DIM + j0 + tid];
        }
    };
    auto store_j = [&](int buf) {           // convert + LDS store
#pragma unroll
        for (int k = 0; k < F4PT; ++k) {
            const int e  = tid + k * NT;
            const int r  = e / (D_DIM / 4);
            const int c4 = e % (D_DIM / 4);
            _Float16* dst = &sEj[buf][r][c4 * 4];
            dst[0] = (_Float16)rj[k].x; dst[1] = (_Float16)rj[k].y;
            dst[2] = (_Float16)rj[k].z; dst[3] = (_Float16)rj[k].w;
        }
        if (tid < TJ) {
            // bounded on causal cells (t_j <= t_ref + tile span): no overflow
            sWj[buf][tid] = beta * __expf(beta * (tjv - tref)) * mjv;
        }
    };
    auto tile_gemm = [&](int buf, int jc0) -> v8f {
        v8f cfrag = {};
#pragma unroll
        for (int cc = 0; cc < 4; ++cc) {
            v16h bfrag;
            v8h* bp = (v8h*)&bfrag;
            bp[0] = *(const v8h*)&sEj[buf][jc0 + n][cc * 32 + hi * 8];
            bp[1] = *(const v8h*)&sEj[buf][jc0 + n][cc * 32 + 16 + hi * 8];
            cfrag = __builtin_amdgcn_wmma_f32_16x16x32_f16(
                false, afrag[cc], false, bfrag,
                (short)0, cfrag, false, false);
        }
        return cfrag;
    };

    // ---- prologue: stage super-tile 0 into buffer 0 ----
    fetch_j(0);
    store_j(0);
    __syncthreads();

    // ---- phase 1: strictly-causal super-tiles, double-buffered ----
    for (int st = 0; st < tile; ++st) {
        const int cur = st & 1;
        fetch_j((st + 1) * TJ);            // global loads in flight ...

        // ... while computing from the current buffer (always exactly 2 tiles)
#pragma unroll
        for (int t = 0; t < 2; ++t) {
            const int jc0 = jpar * 16 + t * 32;
            const v8f cfrag = tile_gemm(cur, jc0);
            const float sw = sWj[cur][jc0 + n];
#pragma unroll
            for (int v = 0; v < 8; ++v)
                racc[v] = fmaf(cfrag[v], sw, racc[v]);
        }

        store_j(cur ^ 1);                  // now wait loads, fill other buffer
        __syncthreads();
    }

    // ---- phase 2: diagonal super-tile (branchless per-element causality) ----
    {
        const int buf = tile & 1;
        const int j0  = tile * TJ;
#pragma unroll
        for (int t = 0; t < 2; ++t) {
            const int jc0 = jpar * 16 + t * 32;
            const v8f cfrag = tile_gemm(buf, jc0);
            const float sw = sWj[buf][jc0 + n];
            const int jg = j0 + jc0 + n;
#pragma unroll
            for (int v = 0; v < 8; ++v) {
                const float w = (jg < igBase + v) ? sw : 0.0f;
                racc[v] = fmaf(cfrag[v], w, racc[v]);
            }
        }
    }

    // ---- apply i-side exp factor, reduce across lanes/waves ----
#pragma unroll
    for (int v = 0; v < 8; ++v) {
        atomicAdd(&sLam[mtile * 16 + v + 8 * hi], racc[v] * eI[v]);
    }
    __syncthreads();

    // ---- lam = rowsum * u ; sum log(lam+eps) ; one atomic per wave ----
    if (tid < TM) {
        const float lam = sLam[tid] * sU[tid];
        float lg = __logf(lam + EPS);
#pragma unroll
        for (int off = 16; off > 0; off >>= 1)
            lg += __shfl_down(lg, off, 32);
        if (lane == 0) atomicAdd(&out[b], lg);
    }
}

extern "C" void kernel_launch(void* const* d_in, const int* in_sizes, int n_in,
                              void* d_out, int out_size, void* d_ws, size_t ws_size,
                              hipStream_t stream) {
    (void)in_sizes; (void)n_in; (void)d_ws; (void)ws_size;
    const int*   ids   = (const int*)  d_in[0];
    const float* times = (const float*)d_in[1];
    const float* maskp = (const float*)d_in[2];
    const float* emb   = (const float*)d_in[3];
    const float* utab  = (const float*)d_in[4];
    const float* betap = (const float*)d_in[5];
    float* out = (float*)d_out;

    hipMemsetAsync(out, 0, (size_t)out_size * sizeof(float), stream);

    dim3 grid(B_DIM * (L_DIM / TM));  // 256 workgroups
    dim3 block(NT);                   // 8 wave32
    mf_hawkes_wmma_kernel<<<grid, block, 0, stream>>>(ids, times, maskp, emb,
                                                      utab, betap, out);
}
